// ContourFormerHead_75539884802139
// MI455X (gfx1250) — matrix operand order repaired and checked by reading, not compiled, not run on hardware
//
#include <hip/hip_runtime.h>

// ---------------------------------------------------------------------------
// ContourFormer head for MI455X (gfx1250): bf16 WMMA GEMMs + fused attention
// GEMM: double-buffered LDS, clamped (branch-free) staging, templated epilogue
// ---------------------------------------------------------------------------
#define B_   4
#define NQ_  300
#define NC_  16
#define C_   256
#define NH_  8
#define NP_  4
#define DFF_ 2048
#define HH_  128
#define WW_  128
#define LQ_  (NQ_ * NC_)      // 4800
#define HD_  (C_ / NH_)       // 32

typedef __attribute__((ext_vector_type(16))) __bf16 v16bf;
typedef __attribute__((ext_vector_type(8)))  __bf16 v8bf;
typedef __attribute__((ext_vector_type(8)))  float  v8f;

union AB16 { v16bf v; v8bf h[2]; };

__device__ __forceinline__ float warp_sum(float v) {
#pragma unroll
  for (int off = 16; off > 0; off >>= 1) v += __shfl_xor(v, off, 32);
  return v;
}

// --------------------------- f32 -> bf16 copy ------------------------------
__global__ void f32_to_bf16_kernel(const float* __restrict__ src,
                                   __bf16* __restrict__ dst, size_t n) {
  size_t i = (size_t)blockIdx.x * blockDim.x + threadIdx.x;
  size_t stride = (size_t)gridDim.x * blockDim.x;
  for (; i < n; i += stride) dst[i] = (__bf16)src[i];
}

// (B,NQ,NC,C) -> (B,NC,NQ,C) transpose + convert
__global__ void sc_transpose_bf16_kernel(const float* __restrict__ sub,
                                         __bf16* __restrict__ dst) {
  size_t n = (size_t)B_ * NQ_ * NC_ * C_;
  size_t i = (size_t)blockIdx.x * blockDim.x + threadIdx.x;
  size_t stride = (size_t)gridDim.x * blockDim.x;
  for (; i < n; i += stride) {
    int c = (int)(i & (C_ - 1));
    size_t t = i >> 8;
    int q = (int)(t % NQ_); t /= NQ_;
    int nc = (int)(t % NC_);
    int b = (int)(t / NC_);
    size_t srcIdx = (((size_t)b * NQ_ + q) * NC_ + nc) * C_ + c;
    dst[i] = (__bf16)sub[srcIdx];
  }
}

// --------------------------- WMMA GEMM -------------------------------------
// D[M,N] = A[M,K] @ W[N,K]^T + bias ; FLAGS: 1=relu, 2=store bf16
// K must be a multiple of 64. Block 256 thr = 8 waves; tile 16(M) x 128(N).
// Out-of-range rows/cols are CLAMPED during staging (their outputs are
// discarded by the bounds-checked store), so staging is branch-free.
template <int FLAGS>
__global__ void __launch_bounds__(256) gemm_bf16_kernel(
    const __bf16* __restrict__ A, const __bf16* __restrict__ W,
    const float* __restrict__ bias, void* __restrict__ Dout,
    int M, int N, int K) {
  __shared__ __attribute__((aligned(32))) __bf16 As[2][16][64];    // [m][k]
  __shared__ __attribute__((aligned(32))) __bf16 Bsw[2][128][64];  // [n][k]
  const int tid  = threadIdx.x;
  const int lane = tid & 31;
  const int wave = tid >> 5;
  const int hi   = lane >> 4;
  const int ln   = lane & 15;
  const int m0   = blockIdx.y * 16;
  const int n0   = blockIdx.x * 128;

  const int ar = tid >> 4;            // A row 0..15
  const int ac = (tid & 15) * 4;      // A col chunk (4 halves = 8B)
  const int br = tid >> 1;            // B row 0..127
  const int bs = (tid & 1) * 32;      // B seg (32 halves = 64B)
  int arow = m0 + ar; if (arow >= M) arow = M - 1;   // clamp, not zero-fill
  int bcol = n0 + br; if (bcol >= N) bcol = N - 1;
  const __bf16* aSrc = A + (size_t)arow * K + ac;
  const __bf16* bSrc = W + (size_t)bcol * K + bs;

  auto stage = [&](int buf, int kt) {
    *(uint2*)&As[buf][ar][ac] = *(const uint2*)(aSrc + kt);
    uint4* dst = (uint4*)&Bsw[buf][br][bs];
    const uint4* s = (const uint4*)(bSrc + kt);
    dst[0] = s[0];
    dst[1] = s[1];
  };

  stage(0, 0);
  __syncthreads();

  v8f acc = {};
  const int nb = wave * 16 + ln;
  int buf = 0;
  for (int kt = 0; kt < K; kt += 64) {
    if (kt + 64 < K) {
      stage(buf ^ 1, kt + 64);                       // overlap with WMMAs
      __builtin_prefetch((const void*)(aSrc + kt + 128), 0, 1);
      __builtin_prefetch((const void*)(bSrc + kt + 128), 0, 1);
    }
#pragma unroll
    for (int kk = 0; kk < 2; ++kk) {
      AB16 a, b;
      a.h[0] = *(const v8bf*)&As[buf][ln][kk * 32 + 16 * hi];
      a.h[1] = *(const v8bf*)&As[buf][ln][kk * 32 + 16 * hi + 8];
      b.v = *(const v16bf*)&Bsw[buf][nb][kk * 32 + 16 * hi];
      acc = __builtin_amdgcn_wmma_f32_16x16x32_bf16(false, a.v, false, b.v,
                                                    (short)0, acc, false, false);
    }
    __syncthreads();
    buf ^= 1;
  }

  int ncol = n0 + wave * 16 + ln;
  if (ncol < N) {
    float bv = bias[ncol];
#pragma unroll
    for (int r = 0; r < 8; ++r) {
      int mrow = m0 + r + 8 * hi;
      if (mrow < M) {
        float v = acc[r] + bv;
        if (FLAGS & 1) v = fmaxf(v, 0.f);
        if (FLAGS & 2) ((__bf16*)Dout)[(size_t)mrow * N + ncol] = (__bf16)v;
        else           ((float*)Dout)[(size_t)mrow * N + ncol] = v;
      }
    }
  }
}

// --------------------- fused attention (one wave / 16 queries) --------------
// qkv: [Bb, L, 3*C] bf16 (q|k|v), out: [Bb, L, C] bf16
__global__ void __launch_bounds__(32) attn_kernel(const __bf16* __restrict__ qkv,
                                                  __bf16* __restrict__ out,
                                                  int L, float scale) {
  __shared__ __attribute__((aligned(32))) float  S[16][320];
  __shared__ __attribute__((aligned(32))) __bf16 P[16][320];
  __shared__ __attribute__((aligned(32))) __bf16 Vt[32][336];  // [dim][key]
  const int lane = threadIdx.x & 31;
  const int hi = lane >> 4, ln = lane & 15;
  const int qt = blockIdx.x, h = blockIdx.y, b = blockIdx.z;
  const int C3 = 3 * C_;

  // ---- stage V transposed: Vt[dim][key], one key column per lane ----
  for (int kb = 0; kb < 10; ++kb) {
    int key = kb * 32 + lane;
    int kc = key < L ? key : L - 1;
    const __bf16* vp = qkv + ((size_t)b * L + kc) * C3 + 2 * C_ + h * HD_;
    v16bf v0 = *(const v16bf*)vp;
    v16bf v1 = *(const v16bf*)(vp + 16);
#pragma unroll
    for (int d = 0; d < 16; ++d) {
      Vt[d][key]      = v0[d];
      Vt[16 + d][key] = v1[d];
    }
  }

  // ---- Q fragment: two contiguous 16B runs ----
  int qr = qt * 16 + ln; if (qr >= L) qr = L - 1;
  const __bf16* qp = qkv + ((size_t)b * L + qr) * C3 + h * HD_;
  AB16 aq;
  aq.h[0] = *(const v8bf*)(qp + 16 * hi);
  aq.h[1] = *(const v8bf*)(qp + 16 * hi + 8);

  // ---- scores: S = Q K^T over 20 key tiles (pad 320) ----
  for (int kt = 0; kt < 20; ++kt) {
    int nk = kt * 16 + ln; if (nk >= L) nk = L - 1;
    const __bf16* kp = qkv + ((size_t)b * L + nk) * C3 + C_ + h * HD_ + 16 * hi;
    v16bf bk = *(const v16bf*)kp;               // 16 contiguous halves (32B)
    v8f c = {};
    c = __builtin_amdgcn_wmma_f32_16x16x32_bf16(false, aq.v, false, bk,
                                                (short)0, c, false, false);
#pragma unroll
    for (int r = 0; r < 8; ++r) S[r + 8 * hi][kt * 16 + ln] = c[r];
  }
  __syncthreads();

  // ---- softmax per query row ----
  if (lane < 16) {
    float mx = -1e30f;
    for (int j = 0; j < L; ++j) {
      float v = S[lane][j] * scale;
      S[lane][j] = v;
      mx = fmaxf(mx, v);
    }
    float sum = 0.f;
    for (int j = 0; j < L; ++j) {
      float e = __expf(S[lane][j] - mx);
      S[lane][j] = e;
      sum += e;
    }
    float inv = 1.f / sum;
    for (int j = 0; j < 320; ++j)
      P[lane][j] = (j < L) ? (__bf16)(S[lane][j] * inv) : (__bf16)0.f;
  }
  __syncthreads();

  // ---- O = P @ V : hd=32 -> two 16-wide n-tiles, 10 K-chunks of 32 keys ----
  v8f a0 = {}, a1 = {};
  for (int kc = 0; kc < 10; ++kc) {
    AB16 ap;
    ap.h[0] = *(const v8bf*)&P[ln][kc * 32 + 16 * hi];
    ap.h[1] = *(const v8bf*)&P[ln][kc * 32 + 16 * hi + 8];
    AB16 b0, b1;
    b0.h[0] = *(const v8bf*)&Vt[ln][kc * 32 + 16 * hi];
    b0.h[1] = *(const v8bf*)&Vt[ln][kc * 32 + 16 * hi + 8];
    b1.h[0] = *(const v8bf*)&Vt[16 + ln][kc * 32 + 16 * hi];
    b1.h[1] = *(const v8bf*)&Vt[16 + ln][kc * 32 + 16 * hi + 8];
    a0 = __builtin_amdgcn_wmma_f32_16x16x32_bf16(false, ap.v, false, b0.v,
                                                 (short)0, a0, false, false);
    a1 = __builtin_amdgcn_wmma_f32_16x16x32_bf16(false, ap.v, false, b1.v,
                                                 (short)0, a1, false, false);
  }
#pragma unroll
  for (int r = 0; r < 8; ++r) {
    int qrow = qt * 16 + r + 8 * hi;
    if (qrow < L) {
      size_t o = ((size_t)b * L + qrow) * C_ + h * HD_;
      out[o + ln]      = (__bf16)a0[r];
      out[o + 16 + ln] = (__bf16)a1[r];
    }
  }
}

// --------------------- combine + LayerNorm ----------------------------------
// tgt = LN(sub + 2*(sc_o + inst_o))  ; one block (256 thr) per row of C=256
__global__ void combine_ln_kernel(const float* __restrict__ sub,
                                  const float* __restrict__ sc_o,
                                  const float* __restrict__ inst_o,
                                  const float* __restrict__ g,
                                  const float* __restrict__ bt,
                                  float* __restrict__ tgt,
                                  __bf16* __restrict__ tgtb) {
  __shared__ float red1[8], red2[8];
  int row = blockIdx.x;               // b*4800 + q*16 + nc
  int t = threadIdx.x;
  int b = row / LQ_, lq = row % LQ_;
  int q = lq >> 4, nc = lq & 15;
  float v = sub[(size_t)row * C_ + t] +
            2.f * (sc_o[(((size_t)b * NC_ + nc) * NQ_ + q) * C_ + t] +
                   inst_o[((size_t)b * NQ_ + q) * C_ + t]);
  float s1 = warp_sum(v), s2 = warp_sum(v * v);
  if ((t & 31) == 0) { red1[t >> 5] = s1; red2[t >> 5] = s2; }
  __syncthreads();
  float mean = 0.f, m2 = 0.f;
#pragma unroll
  for (int w = 0; w < 8; ++w) { mean += red1[w]; m2 += red2[w]; }
  mean *= (1.f / C_);
  float var = m2 * (1.f / C_) - mean * mean;
  float o = (v - mean) * rsqrtf(var + 1e-5f) * g[t] + bt[t];
  tgt[(size_t)row * C_ + t] = o;
  tgtb[(size_t)row * C_ + t] = (__bf16)o;
}

// tgt = LN(x + y)
__global__ void add_ln_kernel(const float* __restrict__ x,
                              const float* __restrict__ y,
                              const float* __restrict__ g,
                              const float* __restrict__ bt,
                              float* __restrict__ outf,
                              __bf16* __restrict__ outb) {
  __shared__ float red1[8], red2[8];
  int row = blockIdx.x, t = threadIdx.x;
  float v = x[(size_t)row * C_ + t] + y[(size_t)row * C_ + t];
  float s1 = warp_sum(v), s2 = warp_sum(v * v);
  if ((t & 31) == 0) { red1[t >> 5] = s1; red2[t >> 5] = s2; }
  __syncthreads();
  float mean = 0.f, m2 = 0.f;
#pragma unroll
  for (int w = 0; w < 8; ++w) { mean += red1[w]; m2 += red2[w]; }
  mean *= (1.f / C_);
  float var = m2 * (1.f / C_) - mean * mean;
  float o = (v - mean) * rsqrtf(var + 1e-5f) * g[t] + bt[t];
  outf[(size_t)row * C_ + t] = o;
  if (outb) outb[(size_t)row * C_ + t] = (__bf16)o;
}

// --------------------- attention-weight softmax (NP=4) ----------------------
__global__ void aw_softmax_kernel(float* __restrict__ aw, int total) {
  int i = blockIdx.x * blockDim.x + threadIdx.x;
  if (i >= total) return;
  float* p = aw + (size_t)i * 4;
  float m = fmaxf(fmaxf(p[0], p[1]), fmaxf(p[2], p[3]));
  float e0 = __expf(p[0] - m), e1 = __expf(p[1] - m);
  float e2 = __expf(p[2] - m), e3 = __expf(p[3] - m);
  float inv = 1.f / (e0 + e1 + e2 + e3);
  p[0] = e0 * inv; p[1] = e1 * inv; p[2] = e2 * inv; p[3] = e3 * inv;
}

// --------------------- deformable bilinear sampling -------------------------
// one block per (b,lq); thread t -> head h=t>>5, channel d=t&31
__global__ void sample_kernel(const float* __restrict__ mem,
                              const float* __restrict__ boxes,
                              const float* __restrict__ off,
                              const float* __restrict__ aw,
                              __bf16* __restrict__ samp) {
  const int row = blockIdx.x;
  const int b = row / LQ_;
  const int t = threadIdx.x;
  const int h = t >> 5, d = t & 31;
  const float cx = boxes[(size_t)row * 4 + 0];
  const float cy = boxes[(size_t)row * 4 + 1];
  const float hw = boxes[(size_t)row * 4 + 2] * 0.5f;
  const float hh = boxes[(size_t)row * 4 + 3] * 0.5f;
  const float* offr = off + (size_t)row * (NH_ * NP_ * 2);
  const float* awr  = aw + (size_t)row * (NH_ * NP_);
  const size_t mb = (size_t)b * (HH_ * WW_) * C_;
  float acc = 0.f;
#pragma unroll
  for (int p = 0; p < NP_; ++p) {
    float lx = cx + offr[(h * NP_ + p) * 2 + 0] * hw;
    float ly = cy + offr[(h * NP_ + p) * 2 + 1] * hh;
    float gx = lx * (float)WW_ - 0.5f;
    float gy = ly * (float)HH_ - 0.5f;
    float x0f = floorf(gx), y0f = floorf(gy);
    int x0 = (int)x0f, y0 = (int)y0f;
    float fx = gx - x0f, fy = gy - y0f;
    float s = 0.f;
#pragma unroll
    for (int dy = 0; dy < 2; ++dy) {
#pragma unroll
      for (int dx = 0; dx < 2; ++dx) {
        int xi = x0 + dx, yi = y0 + dy;
        float wgt = (dx ? fx : 1.f - fx) * (dy ? fy : 1.f - fy);
        float valid = (xi >= 0 && xi < WW_ && yi >= 0 && yi < HH_) ? 1.f : 0.f;
        int xc = min(max(xi, 0), WW_ - 1), yc = min(max(yi, 0), HH_ - 1);
        s += wgt * valid * mem[mb + ((size_t)yc * WW_ + xc) * C_ + h * HD_ + d];
      }
    }
    acc += awr[h * NP_ + p] * s;
  }
  samp[(size_t)row * C_ + t] = (__bf16)acc;
}

// ---------------------------------------------------------------------------
extern "C" void kernel_launch(void* const* d_in, const int* in_sizes, int n_in,
                              void* d_out, int out_size, void* d_ws, size_t ws_size,
                              hipStream_t stream) {
  (void)in_sizes; (void)n_in; (void)out_size; (void)ws_size;
  const float* inst_q = (const float*)d_in[0];
  const float* sub_q  = (const float*)d_in[1];
  const float* memf   = (const float*)d_in[2];
  const float* boxes  = (const float*)d_in[3];
  const float* inst_in_w = (const float*)d_in[4];
  const float* inst_in_b = (const float*)d_in[5];
  const float* inst_out_w = (const float*)d_in[6];
  const float* inst_out_b = (const float*)d_in[7];
  const float* sc_in_w = (const float*)d_in[8];
  const float* sc_in_b = (const float*)d_in[9];
  const float* sc_out_w = (const float*)d_in[10];
  const float* sc_out_b = (const float*)d_in[11];
  const float* off_w = (const float*)d_in[12];
  const float* off_b = (const float*)d_in[13];
  const float* aw_w = (const float*)d_in[14];
  const float* aw_b = (const float*)d_in[15];
  const float* outp_w = (const float*)d_in[16];
  const float* outp_b = (const float*)d_in[17];
  const float* lin1_w = (const float*)d_in[18];
  const float* lin1_b = (const float*)d_in[19];
  const float* lin2_w = (const float*)d_in[20];
  const float* lin2_b = (const float*)d_in[21];
  const float* n1_g = (const float*)d_in[22];
  const float* n1_b = (const float*)d_in[23];
  const float* n2_g = (const float*)d_in[24];
  const float* n2_b = (const float*)d_in[25];
  const float* n3_g = (const float*)d_in[26];
  const float* n3_b = (const float*)d_in[27];

  const int M_inst = B_ * NQ_;          // 1200
  const int M_sc   = B_ * NC_ * NQ_;    // 19200 (== B*Lq)
  const int M_lq   = B_ * LQ_;          // 19200

  char* wp = (char*)d_ws;
  auto alloc = [&](size_t bytes) -> void* {
    void* r = (void*)wp;
    wp += (bytes + 255) & ~(size_t)255;
    return r;
  };
  __bf16* wb_inst_in  = (__bf16*)alloc((size_t)3 * C_ * C_ * 2);
  __bf16* wb_inst_out = (__bf16*)alloc((size_t)C_ * C_ * 2);
  __bf16* wb_sc_in    = (__bf16*)alloc((size_t)3 * C_ * C_ * 2);
  __bf16* wb_sc_out   = (__bf16*)alloc((size_t)C_ * C_ * 2);
  __bf16* wb_off      = (__bf16*)alloc((size_t)NH_ * NP_ * 2 * C_ * 2);
  __bf16* wb_aw       = (__bf16*)alloc((size_t)NH_ * NP_ * C_ * 2);
  __bf16* wb_outp     = (__bf16*)alloc((size_t)C_ * C_ * 2);
  __bf16* wb_lin1     = (__bf16*)alloc((size_t)DFF_ * C_ * 2);
  __bf16* wb_lin2     = (__bf16*)alloc((size_t)C_ * DFF_ * 2);
  __bf16* x_inst   = (__bf16*)alloc((size_t)M_inst * C_ * 2);
  __bf16* x_sc     = (__bf16*)alloc((size_t)M_sc * C_ * 2);
  __bf16* qkv_inst = (__bf16*)alloc((size_t)M_inst * 3 * C_ * 2);
  __bf16* qkv_sc   = (__bf16*)alloc((size_t)M_sc * 3 * C_ * 2);
  __bf16* ao_inst  = (__bf16*)alloc((size_t)M_inst * C_ * 2);
  __bf16* ao_sc    = (__bf16*)alloc((size_t)M_sc * C_ * 2);
  float* inst_o = (float*)alloc((size_t)M_inst * C_ * 4);
  float* sc_o   = (float*)alloc((size_t)M_sc * C_ * 4);
  float* tgt_f  = (float*)alloc((size_t)M_lq * C_ * 4);
  __bf16* tgt_b = (__bf16*)alloc((size_t)M_lq * C_ * 2);
  float* off_f  = (float*)alloc((size_t)M_lq * NH_ * NP_ * 2 * 4);
  float* aw_f   = (float*)alloc((size_t)M_lq * NH_ * NP_ * 4);
  __bf16* samp_b = (__bf16*)alloc((size_t)M_lq * C_ * 2);
  float* tgt2_f = (float*)alloc((size_t)M_lq * C_ * 4);
  __bf16* hid_b = (__bf16*)alloc((size_t)M_lq * DFF_ * 2);

  auto cvt = [&](const float* s, __bf16* d, size_t n) {
    int blocks = (int)((n + 2047) / 2048);
    if (blocks > 4096) blocks = 4096;
    f32_to_bf16_kernel<<<blocks, 256, 0, stream>>>(s, d, n);
  };
  cvt(inst_in_w, wb_inst_in, (size_t)3 * C_ * C_);
  cvt(inst_out_w, wb_inst_out, (size_t)C_ * C_);
  cvt(sc_in_w, wb_sc_in, (size_t)3 * C_ * C_);
  cvt(sc_out_w, wb_sc_out, (size_t)C_ * C_);
  cvt(off_w, wb_off, (size_t)NH_ * NP_ * 2 * C_);
  cvt(aw_w, wb_aw, (size_t)NH_ * NP_ * C_);
  cvt(outp_w, wb_outp, (size_t)C_ * C_);
  cvt(lin1_w, wb_lin1, (size_t)DFF_ * C_);
  cvt(lin2_w, wb_lin2, (size_t)C_ * DFF_);
  cvt(inst_q, x_inst, (size_t)M_inst * C_);
  sc_transpose_bf16_kernel<<<4096, 256, 0, stream>>>(sub_q, x_sc);

  const float scale = 0.17677669529663687f;  // 1/sqrt(32)

  // QKV projections (bf16 out)
  gemm_bf16_kernel<2><<<dim3(6, (M_inst + 15) / 16), 256, 0, stream>>>(
      x_inst, wb_inst_in, inst_in_b, qkv_inst, M_inst, 3 * C_, C_);
  gemm_bf16_kernel<2><<<dim3(6, (M_sc + 15) / 16), 256, 0, stream>>>(
      x_sc, wb_sc_in, sc_in_b, qkv_sc, M_sc, 3 * C_, C_);

  // fused attention
  attn_kernel<<<dim3(19, NH_, B_), 32, 0, stream>>>(qkv_inst, ao_inst, NQ_, scale);
  attn_kernel<<<dim3(19, NH_, B_ * NC_), 32, 0, stream>>>(qkv_sc, ao_sc, NQ_, scale);

  // output projections (f32 out)
  gemm_bf16_kernel<0><<<dim3(2, (M_inst + 15) / 16), 256, 0, stream>>>(
      ao_inst, wb_inst_out, inst_out_b, inst_o, M_inst, C_, C_);
  gemm_bf16_kernel<0><<<dim3(2, (M_sc + 15) / 16), 256, 0, stream>>>(
      ao_sc, wb_sc_out, sc_out_b, sc_o, M_sc, C_, C_);

  // tgt = LN(sub + 2*(sc_o + inst_o))
  combine_ln_kernel<<<M_lq, 256, 0, stream>>>(sub_q, sc_o, inst_o,
                                              n1_g, n1_b, tgt_f, tgt_b);

  // offsets / attention weights
  gemm_bf16_kernel<0><<<dim3(1, (M_lq + 15) / 16), 256, 0, stream>>>(
      tgt_b, wb_off, off_b, off_f, M_lq, NH_ * NP_ * 2, C_);
  gemm_bf16_kernel<0><<<dim3(1, (M_lq + 15) / 16), 256, 0, stream>>>(
      tgt_b, wb_aw, aw_b, aw_f, M_lq, NH_ * NP_, C_);
  aw_softmax_kernel<<<(M_lq * NH_ + 255) / 256, 256, 0, stream>>>(aw_f, M_lq * NH_);

  // deformable bilinear sampling + weighted sum
  sample_kernel<<<M_lq, 256, 0, stream>>>(memf, boxes, off_f, aw_f, samp_b);

  // tgt2 = samp @ outp_w^T ; tgt = LN(tgt + tgt2)
  gemm_bf16_kernel<0><<<dim3(2, (M_lq + 15) / 16), 256, 0, stream>>>(
      samp_b, wb_outp, outp_b, tgt2_f, M_lq, C_, C_);
  add_ln_kernel<<<M_lq, 256, 0, stream>>>(tgt_f, tgt2_f, n2_g, n2_b, tgt_f, tgt_b);

  // FFN
  gemm_bf16_kernel<3><<<dim3(16, (M_lq + 15) / 16), 256, 0, stream>>>(
      tgt_b, wb_lin1, lin1_b, hid_b, M_lq, DFF_, C_);
  gemm_bf16_kernel<0><<<dim3(2, (M_lq + 15) / 16), 256, 0, stream>>>(
      hid_b, wb_lin2, lin2_b, tgt2_f, M_lq, C_, DFF_);
  add_ln_kernel<<<M_lq, 256, 0, stream>>>(tgt_f, tgt2_f, n3_g, n3_b,
                                          (float*)d_out, nullptr);
}
